// Attention_23785528886098
// MI455X (gfx1250) — matrix operand order repaired and checked by reading, not compiled, not run on hardware
//
#include <hip/hip_runtime.h>
#include <hip/hip_bf16.h>

#define B_  4
#define S_  2048
#define D_  1024
#define H_  16
#define DH_ 64

typedef __attribute__((ext_vector_type(16))) __bf16 v16bf;
typedef __attribute__((ext_vector_type(8)))  __bf16 v8bf;
typedef __attribute__((ext_vector_type(4)))  __bf16 v4bf;
typedef __attribute__((ext_vector_type(8)))  float  v8f;

union V16U { v16bf v; v8bf h[2]; };

#ifdef __has_builtin
#  if __has_builtin(__builtin_amdgcn_tensor_load_to_lds)
#    define HAS_TDM 1
#  else
#    define HAS_TDM 0
#  endif
#else
#  define HAS_TDM 0
#endif

static __device__ __forceinline__ __bf16 f2bf(float f) {
  unsigned u = __builtin_bit_cast(unsigned, f);
  unsigned r = (u + 0x7FFFu + ((u >> 16) & 1u)) >> 16;
  unsigned short s = (unsigned short)r;
  return __builtin_bit_cast(__bf16, s);
}

static __device__ __forceinline__ v8f wmma_bf16(v16bf a, v16bf b, v8f c) {
  // D = A(16x32 bf16) * B(32x16 bf16) + C(16x16 f32)
  return __builtin_amdgcn_wmma_f32_16x16x32_bf16(false, a, false, b, (short)0, c,
                                                 false, false);
}

// A-fragment (16x32 bf16): lane m+16g -> K = 16g..16g+15 (elems 0-15)
static __device__ __forceinline__ v16bf load_afrag(const __bf16* rowp, int g) {
  V16U a;
  a.h[0] = *(const v8bf*)(rowp + 16 * g);
  a.h[1] = *(const v8bf*)(rowp + 16 * g + 8);
  return a.v;
}

#if HAS_TDM
typedef __attribute__((ext_vector_type(4))) unsigned v4u;
typedef __attribute__((ext_vector_type(8))) int      v8i;
typedef __attribute__((ext_vector_type(4))) int      v4i;

// Issue a 2D TENSOR_LOAD_TO_LDS with a hand-built D# (bf16 elements).
// Rows of tile_d1 x tile_d0 elements; optional LDS padding every (1<<pi)*8
// bytes of (pa+1) DWORDs to spread ds banks.
static __device__ __forceinline__ void tdm_load_2d(
    const __bf16* gsrc, unsigned lds_off,
    unsigned tensor_d0, unsigned tensor_d1, unsigned stride0,
    unsigned tile_d0, unsigned tile_d1, unsigned pi, unsigned pa) {
  unsigned long long ga = (unsigned long long)(size_t)gsrc;
  v4u g0;
  g0[0] = 1u;                                           // count=1 (valid), user mode
  g0[1] = lds_off;                                      // lds_addr (bytes)
  g0[2] = (unsigned)(ga & 0xFFFFFFFFu);                 // global_addr[31:0]
  g0[3] = (unsigned)((ga >> 32) & 0x01FFFFFFu) | (2u << 30);  // [56:32] | type=2
  v8i g1;
  g1[0] = (int)((1u << 16) | (1u << 20) | (pi << 22) | (pa << 25));
  //            data_size=2B  pad_enable   pad_interval   pad_amount
  g1[1] = (int)((tensor_d0 & 0xFFFFu) << 16);           // tensor_dim0[15:0]
  g1[2] = (int)(((tensor_d0 >> 16) & 0xFFFFu) | ((tensor_d1 & 0xFFFFu) << 16));
  g1[3] = (int)(((tensor_d1 >> 16) & 0xFFFFu) | ((tile_d0 & 0xFFFFu) << 16));
  g1[4] = (int)(tile_d1 & 0xFFFFu);                     // tile_dim1; tile_dim2=0
  g1[5] = (int)stride0;                                 // tensor_dim0_stride[31:0]
  g1[6] = 0;
  g1[7] = 0;
  v4i z = {0, 0, 0, 0};
#if __clang_major__ >= 23
  v8i z8 = {0, 0, 0, 0, 0, 0, 0, 0};
  __builtin_amdgcn_tensor_load_to_lds(g0, g1, z, z, z8, 0);
#else
  __builtin_amdgcn_tensor_load_to_lds(g0, g1, z, z, 0);
#endif
}
#endif  // HAS_TDM

// ---------------------------------------------------------------- conversion
__global__ __launch_bounds__(256) void k_f32_to_bf16(const float* __restrict__ in,
                                                     __bf16* __restrict__ out, int n) {
  int i = (blockIdx.x * blockDim.x + threadIdx.x) * 4;
  if (i < n) {
    float4 f = *(const float4*)(in + i);
    v4bf o;
    o[0] = f2bf(f.x); o[1] = f2bf(f.y); o[2] = f2bf(f.z); o[3] = f2bf(f.w);
    *(v4bf*)(out + i) = o;
  }
}

// ---------------------------------------------------------------- QKV proj
// 2x2 register-blocked: each wave computes a 32x32 tile (4 WMMAs / K-step).
// q/k stored [b,h,p,e]; v stored transposed [b,h,e,p].
__global__ __launch_bounds__(256) void k_qkv_proj(
    const __bf16* __restrict__ xb,
    const __bf16* __restrict__ wq, const __bf16* __restrict__ wk,
    const __bf16* __restrict__ wv,
    __bf16* __restrict__ qb, __bf16* __restrict__ kb, __bf16* __restrict__ vtb) {
  const int lane = threadIdx.x & 31;
  const int wid  = threadIdx.x >> 5;
  const int m = lane & 15, g = lane >> 4;
  const int row0  = blockIdx.x * 128 + (wid >> 1) * 32;
  const int ncol0 = (wid & 1) * 32;
  const int b = blockIdx.y / H_, h = blockIdx.y % H_;
  const int which = blockIdx.z;                          // 0=Q 1=K 2=V
  const __bf16* W = which == 0 ? wq : (which == 1 ? wk : wv);

  const __bf16* arow0 = xb + ((size_t)b * S_ + row0 + m) * D_;
  const __bf16* arow1 = arow0 + (size_t)16 * D_;
  const __bf16* brow0 = W + ((size_t)h * DH_ + ncol0 + m) * D_;
  const __bf16* brow1 = brow0 + (size_t)16 * D_;

  v8f c00 = {}, c01 = {}, c10 = {}, c11 = {};
  for (int k = 0; k < D_; k += 32) {
    v16bf a0 = load_afrag(arow0 + k, g);
    v16bf a1 = load_afrag(arow1 + k, g);
    v16bf b0 = *(const v16bf*)(brow0 + k + 16 * g);
    v16bf b1 = *(const v16bf*)(brow1 + k + 16 * g);
    c00 = wmma_bf16(a0, b0, c00);
    c01 = wmma_bf16(a0, b1, c01);
    c10 = wmma_bf16(a1, b0, c10);
    c11 = wmma_bf16(a1, b1, c11);
  }

  const size_t bh = (size_t)b * H_ + h;
  v8f acc[2][2] = {{c00, c01}, {c10, c11}};
  if (which == 2) {  // V transposed [b,h,e,p]: contiguous over p
#pragma unroll
    for (int i = 0; i < 2; ++i)
#pragma unroll
      for (int j = 0; j < 2; ++j) {
        size_t idx = (bh * DH_ + ncol0 + 16 * j + m) * S_ + row0 + 16 * i + 8 * g;
        v8bf vv;
#pragma unroll
        for (int r = 0; r < 8; ++r) vv[r] = f2bf(acc[i][j][r]);
        *(v8bf*)(vtb + idx) = vv;
      }
  } else {
    __bf16* out = which == 0 ? qb : kb;
#pragma unroll
    for (int i = 0; i < 2; ++i)
#pragma unroll
      for (int j = 0; j < 2; ++j)
#pragma unroll
        for (int r = 0; r < 8; ++r)
          out[(bh * S_ + row0 + 16 * i + r + 8 * g) * DH_ + ncol0 + 16 * j + m] =
              f2bf(acc[i][j][r]);
  }
}

// ---------------------------------------------------------------- attention
// Each wave owns 32 query rows (2 m-tiles). K/V tiles are streamed into LDS by
// the Tensor Data Mover, double-buffered so the DMA of key-block j+1 overlaps
// the WMMA + online-softmax work on block j.
#if HAS_TDM
#define KSTR 72   // 64 dh elems + 8 pad (TDM pad: 16B per 128B row)
#define VSTR 40   // 32 key elems + 8 pad (TDM pad: 16B per 64B row)
#endif

__global__ __launch_bounds__(128) void k_attention(
    const __bf16* __restrict__ qb, const __bf16* __restrict__ kb,
    const __bf16* __restrict__ vtb, __bf16* __restrict__ zb) {
#if HAS_TDM
  __shared__ __bf16 ldsK[4][2][32 * KSTR];   // per-wave, double-buffered K tile
  __shared__ __bf16 ldsV[4][2][64 * VSTR];   // per-wave, double-buffered V tile
#endif
  __shared__ __bf16 ldsP[4][32 * 32];        // per-wave 32x32 P tile
  const int lane = threadIdx.x & 31;
  const int wid  = threadIdx.x >> 5;
  const int m = lane & 15, g = lane >> 4;
  const int qrow0 = (blockIdx.x * 4 + wid) * 32;
  const int b = blockIdx.y / H_, h = blockIdx.y % H_;
  const size_t bh = (size_t)b * H_ + h;

  // Preload Q A-fragments: [m-tile][K chunk of 32 within dh=64]
  V16U qa[2][2];
#pragma unroll
  for (int mt = 0; mt < 2; ++mt) {
    const size_t qrow = (bh * S_ + qrow0 + 16 * mt + m) * DH_;
#pragma unroll
    for (int kk = 0; kk < 2; ++kk) {
      qa[mt][kk].h[0] = *(const v8bf*)(qb + qrow + kk * 32 + 16 * g);
      qa[mt][kk].h[1] = *(const v8bf*)(qb + qrow + kk * 32 + 16 * g + 8);
    }
  }

  v8f o[2][4] = {};
  float mr[2][8], lr[2][8];
#pragma unroll
  for (int mt = 0; mt < 2; ++mt)
#pragma unroll
    for (int r = 0; r < 8; ++r) { mr[mt][r] = -1e30f; lr[mt][r] = 0.f; }

  __bf16* L = ldsP[wid];
  const float scale = 0.125f;  // 1/sqrt(64)
  const int nblk = qrow0 / 32 + 1;

#if HAS_TDM
  auto stage = [&](int jb2, int bf) {
    // K tile: 32 keys x 64 dh, rows contiguous (stride 64 elems in memory)
    tdm_load_2d(kb + (bh * S_ + jb2) * DH_,
                (unsigned)(size_t)&ldsK[wid][bf][0],
                /*t_d0*/64, /*t_d1*/32, /*stride0*/64,
                /*tile_d0*/64, /*tile_d1*/32, /*pi*/4, /*pa*/3);
    // V tile (transposed layout): 64 dh rows x 32 keys, row stride S_
    tdm_load_2d(vtb + bh * (size_t)DH_ * S_ + jb2,
                (unsigned)(size_t)&ldsV[wid][bf][0],
                /*t_d0*/32, /*t_d1*/64, /*stride0*/S_,
                /*tile_d0*/32, /*tile_d1*/64, /*pi*/3, /*pa*/3);
  };
  stage(0, 0);
#endif

  for (int ib = 0; ib < nblk; ++ib) {
    const int jb = ib * 32;
#if HAS_TDM
    const int buf = ib & 1;
    asm volatile("s_wait_tensorcnt 0x0" ::: "memory");  // current tiles resident
    if (ib + 1 < nblk) {
      asm volatile("s_wait_dscnt 0x0" ::: "memory");    // prior reads of buf^1 done
      stage(jb + 32, buf ^ 1);                          // overlap DMA with compute
    }
    const __bf16* Kt = &ldsK[wid][buf][0];
    const __bf16* Vt = &ldsV[wid][buf][0];
#else
    if (jb + 32 < qrow0 + 32) {
      __builtin_prefetch(kb + (bh * S_ + jb + 32 + m) * DH_, 0, 0);
      __builtin_prefetch(vtb + (bh * DH_ + m) * S_ + jb + 32, 0, 0);
    }
#endif

    v8f s[2][2] = {};                                   // [m-tile][key subtile]
#pragma unroll
    for (int kk = 0; kk < 2; ++kk) {
#if HAS_TDM
      v16bf b0 = *(const v16bf*)(Kt + m * KSTR        + kk * 32 + 16 * g);
      v16bf b1 = *(const v16bf*)(Kt + (16 + m) * KSTR + kk * 32 + 16 * g);
#else
      v16bf b0 = *(const v16bf*)(kb + (bh * S_ + jb + m)      * DH_ + kk * 32 + 16 * g);
      v16bf b1 = *(const v16bf*)(kb + (bh * S_ + jb + 16 + m) * DH_ + kk * 32 + 16 * g);
#endif
#pragma unroll
      for (int mt = 0; mt < 2; ++mt) {
        s[mt][0] = wmma_bf16(qa[mt][kk].v, b0, s[mt][0]);
        s[mt][1] = wmma_bf16(qa[mt][kk].v, b1, s[mt][1]);
      }
    }

#pragma unroll
    for (int mt = 0; mt < 2; ++mt) {
      float p0[8], p1[8], alpha[8];
#pragma unroll
      for (int r = 0; r < 8; ++r) {
        int row = qrow0 + 16 * mt + r + 8 * g;          // C layout: M = r+8g, N = m
        float v0 = s[mt][0][r] * scale, v1 = s[mt][1][r] * scale;
        if (jb + m      > row) v0 = -1e30f;             // causal mask
        if (jb + 16 + m > row) v1 = -1e30f;
        float rm = fmaxf(v0, v1);
#pragma unroll
        for (int off = 1; off < 16; off <<= 1)
          rm = fmaxf(rm, __shfl_xor(rm, off, 32));      // row max in half-wave
        float mn = fmaxf(mr[mt][r], rm);
        alpha[r] = __expf(mr[mt][r] - mn);
        mr[mt][r] = mn;
        p0[r] = __expf(v0 - mn);
        p1[r] = __expf(v1 - mn);
        float rs = p0[r] + p1[r];
#pragma unroll
        for (int off = 1; off < 16; off <<= 1)
          rs += __shfl_xor(rs, off, 32);
        lr[mt][r] = lr[mt][r] * alpha[r] + rs;
      }
#pragma unroll
      for (int t = 0; t < 4; ++t)
#pragma unroll
        for (int r = 0; r < 8; ++r) o[mt][t][r] *= alpha[r];

      // P m-tile -> LDS row-major (rows 16*mt.., 32 cols)
#pragma unroll
      for (int r = 0; r < 8; ++r) {
        L[(16 * mt + r + 8 * g) * 32 + m]      = f2bf(p0[r]);
        L[(16 * mt + r + 8 * g) * 32 + 16 + m] = f2bf(p1[r]);
      }
    }
    asm volatile("s_wait_dscnt 0x0" ::: "memory");      // wave-local LDS RAW ordering

    V16U pa[2];
#pragma unroll
    for (int mt = 0; mt < 2; ++mt) {
      pa[mt].h[0] = *(const v8bf*)(L + (16 * mt + m) * 32 + 16 * g);
      pa[mt].h[1] = *(const v8bf*)(L + (16 * mt + m) * 32 + 16 * g + 8);
    }

#pragma unroll
    for (int t = 0; t < 4; ++t) {
#if HAS_TDM
      v16bf vb = *(const v16bf*)(Vt + (t * 16 + m) * VSTR + 16 * g);
#else
      v16bf vb = *(const v16bf*)(vtb + (bh * DH_ + t * 16 + m) * S_ + jb + 16 * g);
#endif
#pragma unroll
      for (int mt = 0; mt < 2; ++mt)
        o[mt][t] = wmma_bf16(pa[mt].v, vb, o[mt][t]);
    }
  }

  // normalize; store z as [b, q, h*64 + e] for the output-projection GEMM
#pragma unroll
  for (int mt = 0; mt < 2; ++mt)
#pragma unroll
    for (int r = 0; r < 8; ++r) {
      float inv = 1.0f / lr[mt][r];
      size_t q = (size_t)qrow0 + 16 * mt + r + 8 * g;
      size_t base = ((size_t)b * S_ + q) * (H_ * DH_) + (size_t)h * DH_;
#pragma unroll
      for (int t = 0; t < 4; ++t)
        zb[base + t * 16 + m] = f2bf(o[mt][t][r] * inv);
    }
}

// ---------------------------------------------------------------- out proj
// 2x2 register-blocked: wave computes 32x32 of out = z @ W_O^T.
__global__ __launch_bounds__(256) void k_out_proj(
    const __bf16* __restrict__ zb, const __bf16* __restrict__ wob,
    float* __restrict__ out) {
  const int lane = threadIdx.x & 31;
  const int wid  = threadIdx.x >> 5;
  const int m = lane & 15, g = lane >> 4;
  const int row0  = blockIdx.x * 128 + (wid >> 1) * 32;   // over B*S rows
  const int ncol0 = blockIdx.y * 64 + (wid & 1) * 32;     // over D cols

  const __bf16* arow0 = zb + ((size_t)row0 + m) * (H_ * DH_);
  const __bf16* arow1 = arow0 + (size_t)16 * (H_ * DH_);
  const __bf16* brow0 = wob + ((size_t)ncol0 + m) * (H_ * DH_);
  const __bf16* brow1 = brow0 + (size_t)16 * (H_ * DH_);

  v8f c00 = {}, c01 = {}, c10 = {}, c11 = {};
  for (int k = 0; k < H_ * DH_; k += 32) {
    v16bf a0 = load_afrag(arow0 + k, g);
    v16bf a1 = load_afrag(arow1 + k, g);
    v16bf b0 = *(const v16bf*)(brow0 + k + 16 * g);
    v16bf b1 = *(const v16bf*)(brow1 + k + 16 * g);
    c00 = wmma_bf16(a0, b0, c00);
    c01 = wmma_bf16(a0, b1, c01);
    c10 = wmma_bf16(a1, b0, c10);
    c11 = wmma_bf16(a1, b1, c11);
  }

  v8f acc[2][2] = {{c00, c01}, {c10, c11}};
#pragma unroll
  for (int i = 0; i < 2; ++i)
#pragma unroll
    for (int j = 0; j < 2; ++j)
#pragma unroll
      for (int r = 0; r < 8; ++r)
        out[((size_t)row0 + 16 * i + r + 8 * g) * D_ + ncol0 + 16 * j + m] =
            acc[i][j][r];
}

// ---------------------------------------------------------------- launcher
extern "C" void kernel_launch(void* const* d_in, const int* in_sizes, int n_in,
                              void* d_out, int out_size, void* d_ws, size_t ws_size,
                              hipStream_t stream) {
  (void)in_sizes; (void)n_in; (void)out_size; (void)ws_size;
  const float* x  = (const float*)d_in[0];
  const float* Wk = (const float*)d_in[1];
  const float* Wq = (const float*)d_in[2];
  const float* Wv = (const float*)d_in[3];
  const float* Wo = (const float*)d_in[4];
  float* out = (float*)d_out;

  char* ws = (char*)d_ws;
  size_t off = 0;
  auto alloc = [&](size_t bytes) -> char* {
    char* p = ws + off;
    off += (bytes + 255) & ~(size_t)255;
    return p;
  };
  __bf16* xb  = (__bf16*)alloc((size_t)B_ * S_ * D_ * 2);        // 16 MB
  __bf16* wkb = (__bf16*)alloc((size_t)H_ * DH_ * D_ * 2);       // 2 MB
  __bf16* wqb = (__bf16*)alloc((size_t)H_ * DH_ * D_ * 2);
  __bf16* wvb = (__bf16*)alloc((size_t)H_ * DH_ * D_ * 2);
  __bf16* wob = (__bf16*)alloc((size_t)D_ * D_ * 2);
  __bf16* qb  = (__bf16*)alloc((size_t)B_ * H_ * S_ * DH_ * 2);  // 16 MB
  __bf16* kb  = (__bf16*)alloc((size_t)B_ * H_ * S_ * DH_ * 2);
  __bf16* vtb = (__bf16*)alloc((size_t)B_ * H_ * S_ * DH_ * 2);  // transposed V
  __bf16* zb  = (__bf16*)alloc((size_t)B_ * S_ * D_ * 2);        // ~92 MB total

  auto conv = [&](const float* src, __bf16* dst, int n) {
    k_f32_to_bf16<<<dim3((n / 4 + 255) / 256), dim3(256), 0, stream>>>(src, dst, n);
  };
  conv(x,  xb,  B_ * S_ * D_);
  conv(Wk, wkb, H_ * DH_ * D_);
  conv(Wq, wqb, H_ * DH_ * D_);
  conv(Wv, wvb, H_ * DH_ * D_);
  conv(Wo, wob, D_ * D_);

  k_qkv_proj<<<dim3(S_ / 128, B_ * H_, 3), dim3(256), 0, stream>>>(
      xb, wqb, wkb, wvb, qb, kb, vtb);
  k_attention<<<dim3(S_ / 128, B_ * H_), dim3(128), 0, stream>>>(qb, kb, vtb, zb);
  k_out_proj<<<dim3((B_ * S_) / 128, D_ / 64), dim3(256), 0, stream>>>(zb, wob, out);
}
